// MobiusGRU_d_38190849196691
// MI455X (gfx1250) — matrix operand order repaired and checked by reading, not compiled, not run on hardware
//
#include <hip/hip_runtime.h>
#include <math.h>

// ---------------------------------------------------------------------------
// Möbius (hyperbolic) 2-layer GRU for MI455X (gfx1250, wave32).
//
// Serial scan of 2048 step-layers -> ONE persistent workgroup per layer
// (grid=2), all recurrent state in LDS (~289 KB of 320 KB/WGP), layer 1
// pipelined behind layer 0 via device-scope flags in L2.  Weights f16 in L2
// (0.75 MB/layer); GEMMs via v_wmma_f32_16x16x32_f16 (fp32 accum).
//
// GEMM structure (512 threads = 16 wave32 waves, 4/SIMD): wave w owns N-tile
// nt=w and all four M-tiles.  K loop split in two halves; each half caches 4
// B-fragments in registers and feeds FOUR independent accumulator chains
// (one per M-tile) -> WMMA->WMMA RAW hazard fully hidden, 4x B reuse.
// A-operands staged as pre-swizzled f16 in LDS (fragment = 2 x ds_load_b128,
// no per-WMMA cvt).  A zero SGPR offset laundered via asm volatile keeps the
// weight loads inside the time loop (blocks the LICM hoist that spilled B
// fragments to scratch) while preserving pointer provenance so the loads
// stay on the global (not flat) path.
// ---------------------------------------------------------------------------

typedef __attribute__((ext_vector_type(16))) _Float16 v16h;
typedef __attribute__((ext_vector_type(8)))  _Float16 v8h;
typedef __attribute__((ext_vector_type(8)))  float    v8f;

#define MIN_NORMF 1e-15f
#define EPSF      1e-7f

#define BB  64
#define TT  1024
#define HH  256
#define NE  (BB * HH)          // 16384 elements per [B,H] tile
#define NTHREADS 512           // 16 waves -> 4 waves/SIMD

__device__ __forceinline__ float tan_k(float u, float sk) {
  float t = fminf(fmaxf(sk * u, -15.f), 15.f);
  return tanhf(t) / sk;
}
__device__ __forceinline__ float artan_k(float u, float sk) {
  float t = fminf(fmaxf(sk * u, -1.f + EPSF), 1.f - EPSF);
  return atanhf(t) / sk;
}
__device__ __forceinline__ float sigmoidf_(float v) {
  return 1.f / (1.f + expf(-v));
}

// ---- element-wise sweep over the [64 x 256] tile (uniform across block) ----
template <typename F>
__device__ __forceinline__ void forEach(F f) {
  for (int i = threadIdx.x; i < NE; i += NTHREADS) f(i >> 8, i & 255, i);
  __syncthreads();
}

// ---- per-row (batch) reductions: wave w handles rows 4w .. 4w+3 ------------
template <typename F>
__device__ __forceinline__ void rowReduce1(float* __restrict__ o0, F f) {
  const int wave = threadIdx.x >> 5, lane = threadIdx.x & 31;
#pragma unroll
  for (int rr = 0; rr < 4; ++rr) {
    const int row = (wave << 2) + rr;
    float s0 = 0.f;
    for (int c = lane; c < HH; c += 32) s0 += f(row, c);
#pragma unroll
    for (int off = 16; off; off >>= 1) s0 += __shfl_down(s0, off, 32);
    if (lane == 0) o0[row] = s0;
  }
  __syncthreads();
}

template <typename F>
__device__ __forceinline__ void rowReduce2(float* __restrict__ o0,
                                           float* __restrict__ o1, F f) {
  const int wave = threadIdx.x >> 5, lane = threadIdx.x & 31;
#pragma unroll
  for (int rr = 0; rr < 4; ++rr) {
    const int row = (wave << 2) + rr;
    float s0 = 0.f, s1 = 0.f;
    for (int c = lane; c < HH; c += 32) {
      float a, b;
      f(row, c, a, b);
      s0 += a; s1 += b;
    }
#pragma unroll
    for (int off = 16; off; off >>= 1) {
      s0 += __shfl_down(s0, off, 32);
      s1 += __shfl_down(s1, off, 32);
    }
    if (lane == 0) { o0[row] = s0; o1[row] = s1; }
  }
  __syncthreads();
}

template <typename F>
__device__ __forceinline__ void rowReduce3(float* __restrict__ o0,
                                           float* __restrict__ o1,
                                           float* __restrict__ o2, F f) {
  const int wave = threadIdx.x >> 5, lane = threadIdx.x & 31;
#pragma unroll
  for (int rr = 0; rr < 4; ++rr) {
    const int row = (wave << 2) + rr;
    float s0 = 0.f, s1 = 0.f, s2 = 0.f;
    for (int c = lane; c < HH; c += 32) {
      float a, b, d;
      f(row, c, a, b, d);
      s0 += a; s1 += b; s2 += d;
    }
#pragma unroll
    for (int off = 16; off; off >>= 1) {
      s0 += __shfl_down(s0, off, 32);
      s1 += __shfl_down(s1, off, 32);
      s2 += __shfl_down(s2, off, 32);
    }
    if (lane == 0) { o0[row] = s0; o1[row] = s1; o2[row] = s2; }
  }
  __syncthreads();
}

// ---- WMMA GEMM: C[64x256] = A[64x256] @ W^T --------------------------------
// 16 waves: wave w owns N-tile nt=w and all four M-tiles.  K loop split in
// two halves; per half, 4 B-fragments cached in registers and reused by four
// independent accumulator chains (one per M-tile).
// AF: (mt, kt, lane) -> v16h A-fragment from pre-swizzled f16 LDS staging
//     (each lane's 16 halves contiguous -> 2 x ds_load_b128).
// B fragment per CDNA5 ISA 7.12.2: lane<16 -> col n, K kb..kb+7, kb+16..kb+23;
//     lane>=16 -> col n, K kb+8..kb+15, kb+24..kb+31 (two 8-half runs per lane).
template <typename AF, typename CS>
__device__ __forceinline__ void gemmAW(AF afrag,
                                       const _Float16* W,
                                       CS cstore) {
  // Laundered zero offset (SGPR): keeps weight loads inside the time loop
  // without destroying W's global-pointer provenance.
  size_t lz = 0;
  asm volatile("" : "+s"(lz));
  const _Float16* Wl = W + lz;
  const int wave = threadIdx.x >> 5, lane = threadIdx.x & 31;
  const int lmod = lane & 15, lhi = lane >> 4;
  const int n = wave * 16 + lmod;   // nt = wave
  v8f acc[4] = {{}, {}, {}, {}};
#pragma unroll
  for (int kh = 0; kh < 2; ++kh) {
    v16h breg[4];
#pragma unroll
    for (int q = 0; q < 4; ++q) {
      const int kt = kh * 4 + q;
      const v8h* wp = (const v8h*)(Wl + (size_t)n * HH + kt * 32 + lhi * 8);
      v8h blo = wp[0], bhi = wp[2];
#pragma unroll
      for (int j = 0; j < 8; ++j) { breg[q][j] = blo[j]; breg[q][8 + j] = bhi[j]; }
    }
#pragma unroll
    for (int q = 0; q < 4; ++q) {
      const int kt = kh * 4 + q;
#pragma unroll
      for (int mt = 0; mt < 4; ++mt) {
        v16h a = afrag(mt, kt, lane);
        acc[mt] = __builtin_amdgcn_wmma_f32_16x16x32_f16(
            false, a, false, breg[q], (short)0, acc[mt], false, false);
      }
    }
  }
#pragma unroll
  for (int mt = 0; mt < 4; ++mt) {
    const int m0 = mt * 16 + lhi * 8;
#pragma unroll
    for (int r = 0; r < 8; ++r) cstore(m0 + r, n, acc[mt][r]);
  }
  __syncthreads();
}

// ---------------------------------------------------------------------------
// Prologue: weights fp32 -> f16 workspace copies; zero pipeline flags.
// ---------------------------------------------------------------------------
__global__ void mgru_prep(const float* __restrict__ wih0,
                          const float* __restrict__ whh0,
                          const float* __restrict__ wih1,
                          const float* __restrict__ whh1,
                          _Float16* __restrict__ w16,
                          int* __restrict__ flags) {
  const int i = blockIdx.x * blockDim.x + threadIdx.x;
  const int WN = 768 * HH;
  if (i < WN) {
    w16[i]          = (_Float16)wih0[i];
    w16[WN + i]     = (_Float16)whh0[i];
    w16[2 * WN + i] = (_Float16)wih1[i];
    w16[3 * WN + i] = (_Float16)whh1[i];
  }
  if (i < TT) flags[i] = 0;
}

// ---------------------------------------------------------------------------
// Persistent scan kernel: block 0 = layer 0, block 1 = layer 1 (pipelined).
// ---------------------------------------------------------------------------
__global__ __launch_bounds__(NTHREADS)
void mgru_scan(const float* __restrict__ x,     // [B,T,D]
               const float* __restrict__ kptr,  // scalar curvature
               const float* __restrict__ h0,    // [2,B,H]
               const float* __restrict__ b0,    // [3,H]
               const float* __restrict__ b1,    // [3,H]
               const _Float16* __restrict__ w16,
               int* __restrict__ flags,
               float* __restrict__ out)         // [T,B,H] out1 then [2,B,H] ht
{
  __shared__ alignas(16) float    sH[NE];   // 64 KB hidden (tangent -> hx)
  __shared__ alignas(16) float    sC[NE];   // 64 KB GEMM out / gate scratch
  __shared__ alignas(16) _Float16 sAh[NE];  // 32 KB swizzled f16: xe / rh
  __shared__ alignas(16) _Float16 sHh[NE];  // 32 KB swizzled f16: hx
  __shared__ alignas(16) _Float16 sUX0[NE]; // 32 KB Ux for current gate
  __shared__ alignas(16) _Float16 sUX1[NE]; // 32 KB Ux_h (candidate)
  __shared__ alignas(16) _Float16 sZ[NE];   // 32 KB z gate
  __shared__ float red0[BB], red1[BB], red2[BB];
  __shared__ float rXN[BB], rHXN[BB], rRHN[BB];

  const int   layer = blockIdx.x;
  const float k  = kptr[0];
  const float sk = sqrtf(fmaxf(-k, 1e-30f));
  const int   WN = 768 * HH;

  const _Float16* Wih = w16 + (size_t)(layer * 2) * WN;
  const _Float16* Whh = w16 + (size_t)(layer * 2 + 1) * WN;
  const _Float16* W_ir  = Wih;                 // rows 0..255 of w_ih
  const _Float16* W_ihh = Wih + 256 * HH;      // rows 256..511
  const _Float16* W_iz  = Wih + 512 * HH;      // rows 512..767
  const _Float16* W_hr  = Whh;
  const _Float16* W_hhh = Whh + 256 * HH;
  const _Float16* W_hz  = Whh + 512 * HH;
  const float* bb  = (layer == 0) ? b0 : b1;
  const float* b_r = bb;
  const float* b_h = bb + HH;
  const float* b_z = bb + 2 * HH;

  float* outbuf = out;                                  // shared out0/out1
  float* htbuf  = out + (size_t)TT * NE + (size_t)layer * NE;

  // init hidden state (tangent space)
  for (int i = threadIdx.x; i < NE; i += NTHREADS) sH[i] = h0[layer * NE + i];
  __syncthreads();

  // ---- swizzled f16 staging stores (match A-fragment layout) ------------
  // element (row m, col kk of 256): mt=m>>4, kt=kk>>5, group g=(kk&31)>>3,
  // lane = ((g&1)<<4) + (m&15), half j = (kk&7) + ((g>>1)<<3).
  auto swzA = [&](int r, int c, float v) {
    int mt = r >> 4, lm = r & 15, kt = c >> 5, kk = c & 31;
    int g = kk >> 3;
    int j = (kk & 7) + ((g >> 1) << 3);
    int lane = ((g & 1) << 4) + lm;
    sAh[(((mt << 3) + kt) * 32 + lane) * 16 + j] = (_Float16)v;
  };
  auto swzH = [&](int r, int c, float v) {
    int mt = r >> 4, lm = r & 15, kt = c >> 5, kk = c & 31;
    int g = kk >> 3;
    int j = (kk & 7) + ((g >> 1) << 3);
    int lane = ((g & 1) << 4) + lm;
    sHh[(((mt << 3) + kt) * 32 + lane) * 16 + j] = (_Float16)v;
  };
  auto fragA = [&](int mt, int kt, int lane) {
    const v8h* p = (const v8h*)&sAh[(((mt << 3) + kt) * 32 + lane) * 16];
    v8h lo = p[0], hi = p[1];
    v16h a;
#pragma unroll
    for (int j = 0; j < 8; ++j) { a[j] = lo[j]; a[8 + j] = hi[j]; }
    return a;
  };
  auto fragH = [&](int mt, int kt, int lane) {
    const v8h* p = (const v8h*)&sHh[(((mt << 3) + kt) * 32 + lane) * 16];
    v8h lo = p[0], hi = p[1];
    v16h a;
#pragma unroll
    for (int j = 0; j < 8; ++j) { a[j] = lo[j]; a[8 + j] = hi[j]; }
    return a;
  };

  auto gemm_A = [&](const _Float16* W) {
    gemmAW(fragA, W, [&](int m, int n, float v) { sC[m * HH + n] = v; });
  };
  auto gemm_H = [&](const _Float16* W) {
    gemmAW(fragH, W, [&](int m, int n, float v) { sC[m * HH + n] = v; });
  };

  // mobius_matvec scaling of sC; nSel: 0=rXN 1=rHXN 2=rRHN;
  // dstSel: 0 -> sC (f32, in place), 1 -> sUX0 (f16), 2 -> sUX1 (f16)
  auto matscale = [&](int nSel, int dstSel) {
    rowReduce1(red0, [&](int r, int c) { float v = sC[r * HH + c]; return v * v; });
    forEach([&](int r, int c, int i) {
      float An  = (nSel == 0) ? rXN[r] : (nSel == 1) ? rHXN[r] : rRHN[r];
      float mxn = fmaxf(sqrtf(red0[r]), MIN_NORMF);
      float s = tan_k((mxn / An) * artan_k(An, sk), sk);
      float v = s * sC[i] / mxn;
      if (dstSel == 0)      sC[i]   = v;
      else if (dstSel == 1) sUX0[i] = (_Float16)v;
      else                  sUX1[i] = (_Float16)v;
    });
  };

  // sC = mobius_add(sC, Y) with Y = sUX0 (0) or sUX1 (1)
  auto madd_ux = [&](int uxSel) {
    rowReduce3(red0, red1, red2, [&](int r, int c, float& a, float& b, float& d) {
      int i = r * HH + c;
      float xv = sC[i];
      float yv = uxSel ? (float)sUX1[i] : (float)sUX0[i];
      a = xv * xv; b = yv * yv; d = xv * yv;
    });
    forEach([&](int r, int c, int i) {
      float x2 = red0[r], y2 = red1[r], xy = red2[r];
      float xv = sC[i];
      float yv = uxSel ? (float)sUX1[i] : (float)sUX0[i];
      float num = (1.f - 2.f * k * xy - k * y2) * xv + (1.f + k * x2) * yv;
      float den = 1.f - 2.f * k * xy + k * k * x2 * y2;
      sC[i] = num / fmaxf(den, MIN_NORMF);
    });
  };

  // sC = mobius_add(sC, bias) (bias broadcast along batch; global f32)
  auto madd_bias = [&](const float* __restrict__ bias) {
    rowReduce3(red0, red1, red2, [&](int r, int c, float& a, float& b, float& d) {
      float xv = sC[r * HH + c], yv = bias[c];
      a = xv * xv; b = yv * yv; d = xv * yv;
    });
    forEach([&](int r, int c, int i) {
      float x2 = red0[r], y2 = red1[r], xy = red2[r];
      float xv = sC[i], yv = bias[c];
      float num = (1.f - 2.f * k * xy - k * y2) * xv + (1.f + k * x2) * yv;
      float den = 1.f - 2.f * k * xy + k * k * x2 * y2;
      sC[i] = num / fmaxf(den, MIN_NORMF);
    });
  };

  for (int t = 0; t < TT; ++t) {
    // ---- acquire input slice -------------------------------------------
    if (layer == 1) {
      if (threadIdx.x == 0) {
        while (__hip_atomic_load(&flags[t], __ATOMIC_ACQUIRE,
                                 __HIP_MEMORY_SCOPE_AGENT) == 0)
          __builtin_amdgcn_s_sleep(2);
      }
      __syncthreads();
      __threadfence();
    }

    // ---- P0: load x_t -> swizzled f16 xe (layer0 applies expmap0) ------
    if (layer == 0) {
      forEach([&](int r, int c, int i) { sC[i] = x[((size_t)r * TT + t) * HH + c]; });
      rowReduce1(red0, [&](int r, int c) { float v = sC[r * HH + c]; return v * v; });
      forEach([&](int r, int c, int i) {
        float un = fmaxf(sqrtf(red0[r]), MIN_NORMF);
        swzA(r, c, tan_k(un, sk) * sC[i] / un);
      });
      if (threadIdx.x < BB) {               // ||xe|| analytically
        float nu = sqrtf(red0[threadIdx.x]);
        float un = fmaxf(nu, MIN_NORMF);
        rXN[threadIdx.x] = fmaxf(tan_k(un, sk) * nu / un, MIN_NORMF);
      }
    } else {
      forEach([&](int r, int c, int i) { sC[i] = outbuf[((size_t)t * BB + r) * HH + c]; });
      rowReduce1(red0, [&](int r, int c) { float v = sC[r * HH + c]; return v * v; });
      forEach([&](int r, int c, int i) { swzA(r, c, sC[i]); });
      if (threadIdx.x < BB)
        rXN[threadIdx.x] = fmaxf(sqrtf(red0[threadIdx.x]), MIN_NORMF);
    }
    __syncthreads();

    // ---- P1: hx = expmap0(h_prev); keep f32 + swizzled f16 copy --------
    rowReduce1(red0, [&](int r, int c) { float v = sH[r * HH + c]; return v * v; });
    forEach([&](int r, int c, int i) {
      float hn = fmaxf(sqrtf(red0[r]), MIN_NORMF);
      float v = tan_k(hn, sk) * sH[i] / hn;
      sH[i] = v;
      swzH(r, c, v);
    });
    if (threadIdx.x < BB) {                 // ||hx|| analytically
      float nh = sqrtf(red0[threadIdx.x]);
      float hn = fmaxf(nh, MIN_NORMF);
      rHXN[threadIdx.x] = fmaxf(tan_k(hn, sk) * nh / hn, MIN_NORMF);
    }
    __syncthreads();

    // ---- P2: Ux_h (candidate's x-side matvec) -> sUX1 ------------------
    gemm_A(W_ihh);  matscale(0, 2);

    // ---- P3: gate z ----------------------------------------------------
    gemm_A(W_iz);   matscale(0, 1);              // Ux_z -> sUX0
    gemm_H(W_hz);   matscale(1, 0);              // Wh_z -> sC
    madd_ux(0);
    madd_bias(b_z);
    rowReduce1(red0, [&](int r, int c) { float v = sC[r * HH + c]; return v * v; });
    forEach([&](int r, int c, int i) {           // logmap0 + sigmoid -> sZ
      float yn = fmaxf(sqrtf(red0[r]), MIN_NORMF);
      sZ[i] = (_Float16)sigmoidf_(artan_k(yn, sk) / yn * sC[i]);
    });

    // ---- P4: gate r ----------------------------------------------------
    gemm_A(W_ir);   matscale(0, 1);              // Ux_r -> sUX0
    gemm_H(W_hr);   matscale(1, 0);              // Wh_r -> sC
    madd_ux(0);
    madd_bias(b_r);
    rowReduce1(red0, [&](int r, int c) { float v = sC[r * HH + c]; return v * v; });
    forEach([&](int r, int c, int i) {           // r gate (f32, in sC)
      float yn = fmaxf(sqrtf(red0[r]), MIN_NORMF);
      sC[i] = sigmoidf_(artan_k(yn, sk) / yn * sC[i]);
    });

    // ---- P5: rh = mobius_pointwise_mul(r, hx) -> swizzled sAh ----------
    rowReduce1(red0, [&](int r, int c) {
      float w = sC[r * HH + c] * sH[r * HH + c];
      return w * w;
    });
    forEach([&](int r, int c, int i) {
      float wxn = fmaxf(sqrtf(red0[r]), MIN_NORMF);
      float s = tan_k((wxn / rHXN[r]) * artan_k(rHXN[r], sk), sk);
      swzA(r, c, s * (sC[i] * sH[i]) / wxn);
    });
    if (threadIdx.x < BB) {                 // ||rh|| analytically
      float nw = sqrtf(red0[threadIdx.x]);
      float wxn = fmaxf(nw, MIN_NORMF);
      float s = tan_k((wxn / rHXN[threadIdx.x]) * artan_k(rHXN[threadIdx.x], sk), sk);
      rRHN[threadIdx.x] = fmaxf(s * nw / wxn, MIN_NORMF);
    }
    __syncthreads();

    // ---- P6: h_tilde = add(add(Wh(rh), Ux_h), b_h) ---------------------
    gemm_A(W_hhh);  matscale(2, 0);              // Wh(rh) -> sC
    madd_ux(1);
    madd_bias(b_h);

    // ---- P7: delta = mobius_add(-hx, h_tilde) -> sC --------------------
    rowReduce3(red0, red1, red2, [&](int r, int c, float& a, float& b, float& d) {
      float xv = -sH[r * HH + c], yv = sC[r * HH + c];
      a = xv * xv; b = yv * yv; d = xv * yv;
    });
    forEach([&](int r, int c, int i) {
      float x2 = red0[r], y2 = red1[r], xy = red2[r];
      float xv = -sH[i], yv = sC[i];
      float num = (1.f - 2.f * k * xy - k * y2) * xv + (1.f + k * x2) * yv;
      float den = 1.f - 2.f * k * xy + k * k * x2 * y2;
      sC[i] = num / fmaxf(den, MIN_NORMF);
    });

    // ---- P8: zdelta = mobius_pointwise_mul(z, delta) -> sC -------------
    rowReduce2(red0, red1, [&](int r, int c, float& a, float& b) {
      int i = r * HH + c;
      float dv = sC[i];
      float wv = (float)sZ[i] * dv;
      a = dv * dv; b = wv * wv;
    });
    forEach([&](int r, int c, int i) {
      float dn  = fmaxf(sqrtf(red0[r]), MIN_NORMF);
      float wxn = fmaxf(sqrtf(red1[r]), MIN_NORMF);
      float s = tan_k((wxn / dn) * artan_k(dn, sk), sk);
      sC[i] = s * ((float)sZ[i] * sC[i]) / wxn;
    });

    // ---- P9: h_new = mobius_add(hx, zdelta) -> sH ----------------------
    rowReduce3(red0, red1, red2, [&](int r, int c, float& a, float& b, float& d) {
      float xv = sH[r * HH + c], yv = sC[r * HH + c];
      a = xv * xv; b = yv * yv; d = xv * yv;
    });
    forEach([&](int r, int c, int i) {
      float x2 = red0[r], y2 = red1[r], xy = red2[r];
      float xv = sH[i], yv = sC[i];
      float num = (1.f - 2.f * k * xy - k * y2) * xv + (1.f + k * x2) * yv;
      float den = 1.f - 2.f * k * xy + k * k * x2 * y2;
      sH[i] = num / fmaxf(den, MIN_NORMF);
    });

    // ---- P10: logmap0 -> carry + global output -------------------------
    rowReduce1(red0, [&](int r, int c) { float v = sH[r * HH + c]; return v * v; });
    forEach([&](int r, int c, int i) {
      float yn = fmaxf(sqrtf(red0[r]), MIN_NORMF);
      float v = artan_k(yn, sk) / yn * sH[i];
      sH[i] = v;
      outbuf[((size_t)t * BB + r) * HH + c] = v;
    });

    // ---- release slice to layer 1 --------------------------------------
    if (layer == 0) {
      __threadfence();
      __syncthreads();
      if (threadIdx.x == 0)
        __hip_atomic_store(&flags[t], 1, __ATOMIC_RELEASE,
                           __HIP_MEMORY_SCOPE_AGENT);
      __syncthreads();
    }
  }

  // final hidden state (tangent) -> ht[layer]
  for (int i = threadIdx.x; i < NE; i += NTHREADS) htbuf[i] = sH[i];
}

// ---------------------------------------------------------------------------
extern "C" void kernel_launch(void* const* d_in, const int* in_sizes, int n_in,
                              void* d_out, int out_size, void* d_ws,
                              size_t ws_size, hipStream_t stream) {
  const float* x    = (const float*)d_in[0];
  const float* k    = (const float*)d_in[1];
  const float* h0   = (const float*)d_in[2];
  const float* wih0 = (const float*)d_in[3];
  const float* whh0 = (const float*)d_in[4];
  const float* b0   = (const float*)d_in[5];
  const float* wih1 = (const float*)d_in[6];
  const float* whh1 = (const float*)d_in[7];
  const float* b1   = (const float*)d_in[8];

  _Float16* w16 = (_Float16*)d_ws;                      // 4 * 768*256 f16
  int* flags = (int*)((char*)d_ws + 4ull * 768 * HH * sizeof(_Float16));
  float* out = (float*)d_out;

  const int WN = 768 * HH;
  mgru_prep<<<(WN + 255) / 256, 256, 0, stream>>>(wih0, whh0, wih1, whh1, w16,
                                                  flags);
  mgru_scan<<<2, NTHREADS, 0, stream>>>(x, k, h0, b0, b1,
                                        (const _Float16*)w16, flags, out);
  (void)in_sizes; (void)n_in; (void)out_size; (void)ws_size;
}
